// Model_36532991819823
// MI455X (gfx1250) — matrix operand order repaired and checked by reading, not compiled, not run on hardware
//
#include <hip/hip_runtime.h>
#include <math.h>

#define DD 256
#define AROW 264   // bf16 elems per staged row (256 + 8 pad -> 528B stride)
#define H1ROW 136  // 128 + 8 pad
#define H2ROW 72   // 64 + 8 pad

typedef __attribute__((ext_vector_type(16))) __bf16          v16bf;
typedef __attribute__((ext_vector_type(8)))  float           v8f;
typedef __attribute__((ext_vector_type(8)))  unsigned short  v8u16;
typedef __attribute__((ext_vector_type(16))) unsigned short  v16u16;

__device__ __forceinline__ unsigned short f32_to_bf16_bits(float f) {
  unsigned u = __float_as_uint(f);
  u += 0x7FFFu + ((u >> 16) & 1u);   // round-to-nearest-even
  return (unsigned short)(u >> 16);
}
__device__ __forceinline__ float bf16_bits_to_f32(unsigned short h) {
  return __uint_as_float(((unsigned)h) << 16);
}
__device__ __forceinline__ v16bf make_frag(v8u16 lo, v8u16 hi) {
  v16u16 u = __builtin_shufflevector(lo, hi, 0,1,2,3,4,5,6,7,8,9,10,11,12,13,14,15);
  return __builtin_bit_cast(v16bf, u);
}
// A fragment (16x32 tile row side): v0-3 = K pairs (k0+hi*8 ..+7), v4-7 = (k0+16+hi*8 ..+7)
__device__ __forceinline__ v16bf load_a_frag(const unsigned short* base, int stride,
                                             int row, int k0, int hi) {
  const v8u16* p0 = (const v8u16*)(base + (size_t)row * stride + k0 + hi * 8);
  const v8u16* p1 = (const v8u16*)(base + (size_t)row * stride + k0 + 16 + hi * 8);
  return make_frag(p0[0], p1[0]);
}
// B fragment (32x16): v0-3 = K (k0+hi*8 ..+7), v4-7 = (k0+16+hi*8 ..+7), same order as A
__device__ __forceinline__ v16bf load_b_frag(const unsigned short* base, int stride,
                                             int col, int k0, int hi) {
  const v8u16* p0 = (const v8u16*)(base + (size_t)col * stride + k0 + hi * 8);
  const v8u16* p1 = (const v8u16*)(base + (size_t)col * stride + k0 + 16 + hi * 8);
  return make_frag(p0[0], p1[0]);
}
#define WMMA_BF16(A, B, C) \
  __builtin_amdgcn_wmma_f32_16x16x32_bf16(false, (A), false, (B), (short)0, (C), false, false)

// ---------------- Prologue: all weights -> bf16 (W3 zero-padded to 208 rows) ----------------
__global__ void prep_kernel(const float* __restrict__ Wk, const float* __restrict__ Wv,
                            const float* __restrict__ Wq, const float* __restrict__ W1,
                            const float* __restrict__ W2, const float* __restrict__ W3,
                            unsigned short* __restrict__ wkv, unsigned short* __restrict__ wq,
                            unsigned short* __restrict__ w1, unsigned short* __restrict__ w2,
                            unsigned short* __restrict__ w3p) {
  int i = blockIdx.x * blockDim.x + threadIdx.x;
  if (i < DD * DD) {
    wkv[i]           = f32_to_bf16_bits(Wk[i]);
    wkv[DD * DD + i] = f32_to_bf16_bits(Wv[i]);
    wq[i]            = f32_to_bf16_bits(Wq[i]);
  }
  if (i < 128 * DD) w1[i] = f32_to_bf16_bits(W1[i]);
  if (i < 64 * 128) w2[i] = f32_to_bf16_bits(W2[i]);
  if (i < 208 * 64) w3p[i] = ((i >> 6) < 200) ? f32_to_bf16_bits(W3[i]) : (unsigned short)0;
}

// ---------------- Gather x_q rows (row seq_len-1 of each batch) as bf16 ----------------
__global__ void gatherq_kernel(const float* __restrict__ input, const int* __restrict__ mask,
                               unsigned short* __restrict__ Xq, int S) {
  __shared__ int slenS;
  const int t = threadIdx.x, b = blockIdx.x;
  if (t == 0) slenS = 0;
  __syncthreads();
  if (t < S) atomicAdd(&slenS, mask[(size_t)b * S + t]);
  __syncthreads();
  int r = slenS - 1; if (r < 0) r = 0;
  Xq[(size_t)b * DD + t] = f32_to_bf16_bits(input[((size_t)b * S + r) * DD + t]);
}

// ---------------- Batched q projection: q = Xq @ Wq^T + bq (WMMA, 64 rows per WG) ----------------
__global__ __launch_bounds__(256, 2)
void qgemm_kernel(const unsigned short* __restrict__ Xq, const unsigned short* __restrict__ wq,
                  const float* __restrict__ bq, float* __restrict__ qout) {
  const int t = threadIdx.x;
  const int rbase = blockIdx.x * 64;
  const int wave = t >> 5, lane = t & 31, lr = lane & 15, hi = lane >> 4;
  const v8f zero = {0.f,0.f,0.f,0.f,0.f,0.f,0.f,0.f};
  for (int ni = 0; ni < 2; ++ni) {
    const int n0 = (wave + ni * 8) * 16;
    const float bias = bq[n0 + lr];
    v8f acc[4];
    #pragma unroll
    for (int m = 0; m < 4; ++m) acc[m] = zero;
    for (int k0 = 0; k0 < DD; k0 += 32) {
      v16bf fb = load_b_frag(wq, DD, n0 + lr, k0, hi);
      #pragma unroll
      for (int mt = 0; mt < 4; ++mt) {
        v16bf fa = load_a_frag(Xq, DD, rbase + mt * 16 + lr, k0, hi);
        acc[mt] = WMMA_BF16(fa, fb, acc[mt]);
      }
    }
    #pragma unroll
    for (int mt = 0; mt < 4; ++mt)
      #pragma unroll
      for (int r = 0; r < 8; ++r)
        qout[(size_t)(rbase + mt * 16 + hi * 8 + r) * DD + n0 + lr] = acc[mt][r] + bias;
  }
}

// ---------------- Fused per-batch attention: K/V WMMA GEMM + softmax + prev ----------------
__global__ __launch_bounds__(256, 2)
void fused_attn(const float* __restrict__ input, const int* __restrict__ mask,
                const float* __restrict__ qglob, const float* __restrict__ bv,
                const unsigned short* __restrict__ wkv,
                unsigned short* __restrict__ xmlp, int S) {
  __shared__ __align__(16) unsigned short Abf[64 * AROW]; // input[b] bf16, zero-padded rows
  __shared__ __align__(16) unsigned short Kl[64 * AROW];  // K (bf16, no bias: drops in softmax)
  __shared__ __align__(16) unsigned short Vl[64 * AROW];  // V (bf16, bv folded later)
  __shared__ float qv[DD], partial[256], scoresS[64], attnS[64];
  __shared__ int slenS;

  const int t = threadIdx.x;
  const int b = blockIdx.x;
  const float* inb = input + (size_t)b * S * DD;

  if (t == 0) slenS = 0;
  __syncthreads();
  if (t < S) atomicAdd(&slenS, mask[(size_t)b * S + t]);
  __syncthreads();
  const int slen = slenS;

  // Stage input rows as bf16 A (rows >= S zeroed); emit nxt row straight to MLP input.
  for (int r = 0; r < 64; ++r) {
    float val = 0.0f;
    if (r < S) val = inb[(size_t)r * DD + t];
    Abf[r * AROW + t] = f32_to_bf16_bits(val);
    if (r == S - 1) xmlp[(size_t)(2 * b + 1) * DD + t] = f32_to_bf16_bits(val);
  }
  qv[t] = qglob[(size_t)b * DD + t];
  __syncthreads();

  // K/V projection GEMM via v_wmma_f32_16x16x32_bf16
  {
    const int wave = t >> 5, lane = t & 31, lr = lane & 15, hi = lane >> 4;
    const unsigned short* WkB = wkv;
    const unsigned short* WvB = wkv + DD * DD;
    const v8f zero = {0.f,0.f,0.f,0.f,0.f,0.f,0.f,0.f};
    for (int ni = 0; ni < 2; ++ni) {
      const int n0 = wave * 32 + ni * 16;
      v8f accK[4], accV[4];
      #pragma unroll
      for (int m = 0; m < 4; ++m) { accK[m] = zero; accV[m] = zero; }
      for (int k0 = 0; k0 < DD; k0 += 32) {
        v16bf fbk = load_b_frag(WkB, DD, n0 + lr, k0, hi);
        v16bf fbv = load_b_frag(WvB, DD, n0 + lr, k0, hi);
        #pragma unroll
        for (int mt = 0; mt < 4; ++mt) {
          v16bf fa = load_a_frag(Abf, AROW, mt * 16 + lr, k0, hi);
          accK[mt] = WMMA_BF16(fa, fbk, accK[mt]);
          accV[mt] = WMMA_BF16(fa, fbv, accV[mt]);
        }
      }
      #pragma unroll
      for (int mt = 0; mt < 4; ++mt)
        #pragma unroll
        for (int r = 0; r < 8; ++r) {
          int row = mt * 16 + hi * 8 + r;
          Kl[row * AROW + n0 + lr] = f32_to_bf16_bits(accK[mt][r]);
          Vl[row * AROW + n0 + lr] = f32_to_bf16_bits(accV[mt][r]);
        }
    }
  }
  __syncthreads();

  // scores[s] = q . K[s]   (bk adds the same constant to every s -> drops out of softmax)
  {
    const int row = t >> 2, p = t & 3;
    float acc = 0.f;
    for (int i = 0; i < 64; ++i) {
      int d = 4 * i + p;
      acc += qv[d] * bf16_bits_to_f32(Kl[row * AROW + d]);
    }
    partial[t] = acc;
  }
  __syncthreads();
  if (t < 64)
    scoresS[t] = partial[4*t] + partial[4*t+1] + partial[4*t+2] + partial[4*t+3];
  __syncthreads();

  // masked softmax over s < slen-1 (50 elements; serial on t0 is negligible)
  if (t == 0) {
    int n = slen - 1;
    float m = -INFINITY;
    for (int s = 0; s < n; ++s) m = fmaxf(m, scoresS[s]);
    float sum = 0.f;
    for (int s = 0; s < n; ++s) { float e = expf(scoresS[s] - m); attnS[s] = e; sum += e; }
    float inv = (sum > 0.f) ? 1.f / sum : 0.f;
    for (int s = 0; s < n; ++s) attnS[s] *= inv;
    for (int s = (n > 0 ? n : 0); s < 64; ++s) attnS[s] = 0.f;
  }
  __syncthreads();

  // prev = tanh(sum_s attn[s]*V[s] + bv)   (sum attn == 1 folds bv); emit as MLP input row 2b
  {
    float acc = 0.f;
    for (int s = 0; s < S; ++s) acc += attnS[s] * bf16_bits_to_f32(Vl[s * AROW + t]);
    xmlp[(size_t)(2 * b) * DD + t] = f32_to_bf16_bits(tanhf(acc + bv[t]));
  }
}

// ---------------- Batched MLP: 3 WMMA stages with fused bias+relu, 64 rows per WG ----------------
__global__ __launch_bounds__(256, 2)
void mlp_kernel(const unsigned short* __restrict__ X,   // (2B x 256) bf16
                const unsigned short* __restrict__ w1, const float* __restrict__ b1,
                const unsigned short* __restrict__ w2, const float* __restrict__ b2,
                const unsigned short* __restrict__ w3, const float* __restrict__ b3,
                float* __restrict__ out) {
  __shared__ __align__(16) unsigned short Xl[64 * AROW];
  __shared__ __align__(16) unsigned short H1[64 * H1ROW];
  __shared__ __align__(16) unsigned short H2[64 * H2ROW];
  const int t = threadIdx.x;
  const int rbase = blockIdx.x * 64;
  for (int r = 0; r < 64; ++r)
    Xl[r * AROW + t] = X[(size_t)(rbase + r) * DD + t];
  __syncthreads();

  const int wave = t >> 5, lane = t & 31, lr = lane & 15, hi = lane >> 4;
  const v8f zero = {0.f,0.f,0.f,0.f,0.f,0.f,0.f,0.f};

  // stage 1: 256 -> 128, relu
  {
    const int n0 = wave * 16;
    const float bias = b1[n0 + lr];
    v8f acc[4];
    #pragma unroll
    for (int m = 0; m < 4; ++m) acc[m] = zero;
    for (int k0 = 0; k0 < 256; k0 += 32) {
      v16bf fb = load_b_frag(w1, 256, n0 + lr, k0, hi);
      #pragma unroll
      for (int mt = 0; mt < 4; ++mt) {
        v16bf fa = load_a_frag(Xl, AROW, mt * 16 + lr, k0, hi);
        acc[mt] = WMMA_BF16(fa, fb, acc[mt]);
      }
    }
    #pragma unroll
    for (int mt = 0; mt < 4; ++mt)
      #pragma unroll
      for (int r = 0; r < 8; ++r) {
        int row = mt * 16 + hi * 8 + r;
        H1[row * H1ROW + n0 + lr] = f32_to_bf16_bits(fmaxf(acc[mt][r] + bias, 0.f));
      }
  }
  __syncthreads();

  // stage 2: 128 -> 64, relu  (waves 0-3)
  for (int nt = wave; nt < 4; nt += 8) {
    const int n0 = nt * 16;
    const float bias = b2[n0 + lr];
    v8f acc[4];
    #pragma unroll
    for (int m = 0; m < 4; ++m) acc[m] = zero;
    for (int k0 = 0; k0 < 128; k0 += 32) {
      v16bf fb = load_b_frag(w2, 128, n0 + lr, k0, hi);
      #pragma unroll
      for (int mt = 0; mt < 4; ++mt) {
        v16bf fa = load_a_frag(H1, H1ROW, mt * 16 + lr, k0, hi);
        acc[mt] = WMMA_BF16(fa, fb, acc[mt]);
      }
    }
    #pragma unroll
    for (int mt = 0; mt < 4; ++mt)
      #pragma unroll
      for (int r = 0; r < 8; ++r) {
        int row = mt * 16 + hi * 8 + r;
        H2[row * H2ROW + n0 + lr] = f32_to_bf16_bits(fmaxf(acc[mt][r] + bias, 0.f));
      }
  }
  __syncthreads();

  // stage 3: 64 -> 200 (N padded to 208; w3 rows 200-207 are zero), relu, store
  for (int nt = wave; nt < 13; nt += 8) {
    const int n0 = nt * 16;
    const int col = n0 + lr;
    const float bias = (col < 200) ? b3[col] : 0.f;
    v8f acc[4];
    #pragma unroll
    for (int m = 0; m < 4; ++m) acc[m] = zero;
    for (int k0 = 0; k0 < 64; k0 += 32) {
      v16bf fb = load_b_frag(w3, 64, col, k0, hi);
      #pragma unroll
      for (int mt = 0; mt < 4; ++mt) {
        v16bf fa = load_a_frag(H2, H2ROW, mt * 16 + lr, k0, hi);
        acc[mt] = WMMA_BF16(fa, fb, acc[mt]);
      }
    }
    if (col < 200) {
      #pragma unroll
      for (int mt = 0; mt < 4; ++mt)
        #pragma unroll
        for (int r = 0; r < 8; ++r) {
          int row = rbase + mt * 16 + hi * 8 + r;   // row = 2b + which
          out[(size_t)row * 200 + col] = fmaxf(acc[mt][r] + bias, 0.f);
        }
    }
  }
}

extern "C" void kernel_launch(void* const* d_in, const int* in_sizes, int n_in,
                              void* d_out, int out_size, void* d_ws, size_t ws_size,
                              hipStream_t stream) {
  const float* input = (const float*)d_in[0];
  const int*   mask  = (const int*)  d_in[1];
  const float* Wq    = (const float*)d_in[2];
  const float* bq    = (const float*)d_in[3];
  const float* Wk    = (const float*)d_in[4];
  // d_in[5] = bk: constant shift of all scores -> drops out of softmax
  const float* Wv    = (const float*)d_in[6];
  const float* bv    = (const float*)d_in[7];
  const float* W1    = (const float*)d_in[8];
  const float* b1    = (const float*)d_in[9];
  const float* W2    = (const float*)d_in[10];
  const float* b2    = (const float*)d_in[11];
  const float* W3    = (const float*)d_in[12];
  const float* b3    = (const float*)d_in[13];

  const int B = out_size / 400;              // out is (B, 2, 200)
  const int S = in_sizes[1] / (B > 0 ? B : 1);

  char* w = (char*)d_ws;                     // workspace layout (bytes)
  unsigned short* wkv = (unsigned short*)(w);            // 262144  (Wk|Wv bf16)
  unsigned short* wqb = (unsigned short*)(w + 262144);   // 131072
  unsigned short* w1b = (unsigned short*)(w + 393216);   // 65536
  unsigned short* w2b = (unsigned short*)(w + 458752);   // 16384
  unsigned short* w3b = (unsigned short*)(w + 475136);   // 26624 (208x64)
  size_t off = 501760;
  unsigned short* Xq  = (unsigned short*)(w + off); off += (size_t)B * DD * 2;
  float*          qf  = (float*)         (w + off); off += (size_t)B * DD * 4;
  unsigned short* Xm  = (unsigned short*)(w + off);      // (2B x 256) bf16

  prep_kernel<<<(DD * DD + 255) / 256, 256, 0, stream>>>(Wk, Wv, Wq, W1, W2, W3,
                                                         wkv, wqb, w1b, w2b, w3b);
  gatherq_kernel<<<B, 256, 0, stream>>>(input, mask, Xq, S);
  qgemm_kernel<<<B / 64, 256, 0, stream>>>(Xq, wqb, bq, qf);
  fused_attn<<<B, 256, 0, stream>>>(input, mask, qf, bv, wkv, Xm, S);
  mlp_kernel<<<(2 * B) / 64, 256, 0, stream>>>(Xm, w1b, b1, w2b, b2, w3b, b3,
                                               (float*)d_out);
}
